// InvariantSlotAttention_disc_69939247448163
// MI455X (gfx1250) — matrix operand order repaired and checked by reading, not compiled, not run on hardware
//
#include <hip/hip_runtime.h>

typedef _Float16 h16;
typedef __attribute__((ext_vector_type(16))) _Float16 v16h;
typedef __attribute__((ext_vector_type(8)))  float    v8f;

#define BB 16
#define SS 7
#define DD 4096
#define EE 64
#define LINC (1.0f/63.0f)

// ---------------- WMMA helpers (CDNA5 16x16x32 f16 -> f32) ----------------

__device__ __forceinline__ v8f wmma16(v16h a, v16h b, v8f c) {
  return __builtin_amdgcn_wmma_f32_16x16x32_f16(false, a, false, b, (short)0, c, false, false);
}

// A fragment (16x32 f16, M x K). Per ISA 7.12.2: lanes 0-15 hold M=0..15 with
// K=0..7 (v0-3) & K=16..23 (v4-7); lanes 16-31 hold K=8..15 & K=24..31.
// Source: row-major [16][64] f16 tile, chunk c selects K-range c*32..c*32+31.
__device__ __forceinline__ v16h load_afrag(const h16* X, int lane, int c) {
  int r = lane & 15, kb = (lane >> 4) * 8;
  union { v16h v; uint4 q[2]; } u;
  const h16* p = X + r * 64 + c * 32 + kb;
  u.q[0] = *(const uint4*)(p);
  u.q[1] = *(const uint4*)(p + 16);
  return u.v;
}

// B fragment (32x16 f16, K x N), pre-shuffled buffer: lane holds 16 contiguous
// halves = B[k][n], n = lane&15, k = (lane>>4)*8 + (t&7) + (t>>3)*16 for half t.
__device__ __forceinline__ v16h load_bfrag(const h16* W, int lane) {
  union { v16h v; uint4 q[2]; } u;
  const uint4* p = (const uint4*)(W + lane * 16);
  u.q[0] = p[0]; u.q[1] = p[1];
  return u.v;
}

__device__ __forceinline__ void lds_wait() {
  asm volatile("s_wait_dscnt 0x0" ::: "memory");
}

__device__ __forceinline__ void async_wait() {
  asm volatile("s_wait_asynccnt 0x0" ::: "memory");
}

// Async global->LDS copy of 16 bytes (tracked on ASYNCcnt, bypasses VGPRs).
__device__ __forceinline__ void async_copy_b128(unsigned lds_off, const void* gptr) {
  unsigned long long ga = (unsigned long long)(size_t)gptr;
  asm volatile("global_load_async_to_lds_b128 %0, %1, off"
               :: "v"(lds_off), "v"(ga) : "memory");
}

// ---------------- weight pre-shuffle into B-fragment layout ----------------

// 64x64 row-major W[k][n] -> frag[(c*4+nc)*512 + lane*16 + t]
__global__ void prep_frag64(const float* __restrict__ W, h16* __restrict__ frag) {
  int idx = blockIdx.x * 256 + threadIdx.x;           // 4096
  int t = idx & 15, lane = (idx >> 4) & 31, g = idx >> 9;
  int c = g >> 2, nc = g & 3;
  int n = nc * 16 + (lane & 15);
  int k = c * 32 + (lane >> 4) * 8 + (t & 7) + (t >> 3) * 16;
  frag[idx] = (h16)W[k * 64 + n];
}

// conv weights [O=64][I=64][5][5] -> frag[tap*4096 + (c*4+nc)*512 + lane*16 + t]
__global__ void prep_conv_frag(const float* __restrict__ W, h16* __restrict__ frag) {
  int idx = blockIdx.x * 256 + threadIdx.x;           // 102400
  int t = idx & 15, lane = (idx >> 4) & 31, g = (idx >> 9) & 7, tap = idx >> 12;
  int c = g >> 2, nc = g & 3;
  int o = nc * 16 + (lane & 15);
  int i = c * 32 + (lane >> 4) * 8 + (t & 7) + (t >> 3) * 16;
  frag[idx] = (h16)W[(o * 64 + i) * 25 + tap];
}

// ---------------- conv0 (1->64ch, VALU f32, writes channel-last f16) --------

__global__ __launch_bounds__(256) void conv0_k(const float* __restrict__ data,
                                               const float* __restrict__ w,
                                               const float* __restrict__ bias,
                                               h16* __restrict__ act0) {
  int idx = blockIdx.x * 256 + threadIdx.x;           // B*D = 65536
  int b = idx >> 12, p = idx & 4095, y = p >> 6, x = p & 63;
  float win[25];
#pragma unroll
  for (int dy = 0; dy < 5; dy++)
#pragma unroll
    for (int dx = 0; dx < 5; dx++) {
      int yy = y + dy - 2, xx = x + dx - 2;
      win[dy * 5 + dx] = (yy >= 0 && yy < 64 && xx >= 0 && xx < 64)
                             ? data[(size_t)b * 4096 + yy * 64 + xx] : 0.0f;
    }
  for (int ob = 0; ob < 64; ob += 8) {
    float a[8];
#pragma unroll
    for (int k = 0; k < 8; k++) a[k] = bias[ob + k];
#pragma unroll
    for (int t = 0; t < 25; t++) {
      float wv = win[t];
#pragma unroll
      for (int k = 0; k < 8; k++) a[k] += wv * w[(ob + k) * 25 + t];
    }
    union { uint4 q; h16 h[8]; } pk;
#pragma unroll
    for (int k = 0; k < 8; k++) pk.h[k] = (h16)fmaxf(a[k], 0.0f);
    *(uint4*)(act0 + (size_t)idx * 64 + ob) = pk.q;
  }
}

// ---------------- conv1/conv2: im2col WMMA GEMM -----------------------------
// Block = 1 batch x 2 image rows (128 pixels), 8 waves x 16-pixel tiles.
// K-loop: 25 taps x 2 chunks of 32 input channels. LN_EPI=1 fuses ReLU+LN.
// Input tile staged via GLOBAL_LOAD_ASYNC_TO_LDS_B128 (ASYNCcnt).

template <int LN_EPI>
__global__ __launch_bounds__(256) void conv5x5(const h16* __restrict__ actin,
                                               const h16* __restrict__ wfrag,
                                               const float* __restrict__ bias,
                                               const float* __restrict__ ng,
                                               const float* __restrict__ nb_,
                                               h16* __restrict__ actout,
                                               float* __restrict__ encout) {
  __shared__ alignas(16) h16 tin[6 * 68 * 64];        // rows y0-2..y0+3, cols padded
  int b = blockIdx.x >> 5, band = blockIdx.x & 31, y0 = band * 2;
  int tid = threadIdx.x;
  unsigned tin_base = (unsigned)(size_t)(void*)tin;
  for (int i = tid; i < 6 * 68 * 8; i += 256) {       // 16-byte chunks (8 halves)
    int ch8 = i & 7, rc = i >> 3, col = rc % 68, row = rc / 68;
    int yy = y0 - 2 + row, xx = col - 2;
    size_t loff = (size_t)(row * 68 + col) * 64 + ch8 * 8;
    if (yy >= 0 && yy < 64 && xx >= 0 && xx < 64) {
      async_copy_b128(tin_base + (unsigned)(loff * 2),
                      actin + ((size_t)b * 4096 + yy * 64 + xx) * 64 + ch8 * 8);
    } else {
      uint4 z = {0u, 0u, 0u, 0u};
      *(uint4*)(tin + loff) = z;
    }
  }
  async_wait();
  __syncthreads();

  int lane = tid & 31, w = tid >> 5, ln15 = lane & 15, hh = lane >> 4;
  int p = w * 16 + ln15;                              // A-row pixel (M = ln15)
  int py_ = p >> 6, px_ = p & 63;
  int kb = hh * 8;

  v8f accv[4] = {v8f{}, v8f{}, v8f{}, v8f{}};
#pragma unroll
  for (int dy = 0; dy < 5; dy++)
#pragma unroll
    for (int dx = 0; dx < 5; dx++) {
      const h16* base = tin + ((py_ + dy) * 68 + (px_ + dx)) * 64;
      union { v16h v; uint4 q[2]; } u0, u1;
      u0.q[0] = *(const uint4*)(base + kb);
      u0.q[1] = *(const uint4*)(base + 16 + kb);
      u1.q[0] = *(const uint4*)(base + 32 + kb);
      u1.q[1] = *(const uint4*)(base + 48 + kb);
      const h16* wf = wfrag + (size_t)(dy * 5 + dx) * 4096;
#pragma unroll
      for (int nc = 0; nc < 4; nc++) {
        accv[nc] = wmma16(u0.v, load_bfrag(wf + nc * 512, lane), accv[nc]);
        accv[nc] = wmma16(u1.v, load_bfrag(wf + (4 + nc) * 512, lane), accv[nc]);
      }
    }

  float bv[4];
#pragma unroll
  for (int nc = 0; nc < 4; nc++) bv[nc] = bias[nc * 16 + ln15];
  size_t pixbase = (size_t)b * 4096 + band * 128 + w * 16;

  if (LN_EPI == 0) {
#pragma unroll
    for (int nc = 0; nc < 4; nc++)
#pragma unroll
      for (int i = 0; i < 8; i++) {
        float v = fmaxf(accv[nc][i] + bv[nc], 0.0f);
        actout[(pixbase + i + 8 * hh) * 64 + nc * 16 + ln15] = (h16)v;
      }
  } else {
    float gv[4], ob[4];
#pragma unroll
    for (int nc = 0; nc < 4; nc++) { gv[nc] = ng[nc * 16 + ln15]; ob[nc] = nb_[nc * 16 + ln15]; }
    float rl[4][8], sm[8], sq[8];
#pragma unroll
    for (int i = 0; i < 8; i++) { sm[i] = 0.0f; sq[i] = 0.0f; }
#pragma unroll
    for (int nc = 0; nc < 4; nc++)
#pragma unroll
      for (int i = 0; i < 8; i++) {
        float v = fmaxf(accv[nc][i] + bv[nc], 0.0f);
        rl[nc][i] = v; sm[i] += v; sq[i] += v * v;
      }
#pragma unroll
    for (int m = 1; m < 16; m <<= 1)
#pragma unroll
      for (int i = 0; i < 8; i++) {
        sm[i] += __shfl_xor(sm[i], m, 32);
        sq[i] += __shfl_xor(sq[i], m, 32);
      }
#pragma unroll
    for (int i = 0; i < 8; i++) {
      float mu = sm[i] * (1.0f / 64.0f);
      float var = sq[i] * (1.0f / 64.0f) - mu * mu;
      float rs = rsqrtf(var + 1e-5f);
      size_t rowb = (pixbase + i + 8 * hh) * 64;
#pragma unroll
      for (int nc = 0; nc < 4; nc++)
        encout[rowb + nc * 16 + ln15] = (rl[nc][i] - mu) * rs * gv[nc] + ob[nc];
    }
  }
}

// ---------------- encK/encV = encLN @ toK_w/toV_w + b (WMMA) ----------------
// B fragments held in registers for the whole kernel (loaded once from L2).

__global__ __launch_bounds__(256) void enc_kv(const float* __restrict__ encLN,
                                              const h16* __restrict__ kfrag,
                                              const h16* __restrict__ vfrag,
                                              const float* __restrict__ kb_,
                                              const float* __restrict__ vb_,
                                              float* __restrict__ encK,
                                              float* __restrict__ encV) {
  __shared__ alignas(16) h16 X[8][1024];
  int tid = threadIdx.x;
  int lane = tid & 31, w = tid >> 5, ln15 = lane & 15, hh = lane >> 4;
  v16h kfr[8], vfr[8];
#pragma unroll
  for (int g = 0; g < 8; g++) {
    kfr[g] = load_bfrag(kfrag + g * 512, lane);
    vfr[g] = load_bfrag(vfrag + g * 512, lane);
  }
  float kbv[4], vbv[4];
#pragma unroll
  for (int nc = 0; nc < 4; nc++) { kbv[nc] = kb_[nc * 16 + ln15]; vbv[nc] = vb_[nc * 16 + ln15]; }
  h16* Xw = X[w];
  for (int t = 0; t < 4; t++) {
    size_t row0 = (size_t)blockIdx.x * 512 + (w * 4 + t) * 16;
    size_t r = row0 + ln15;
    const float4* s4 = (const float4*)(encLN + r * 64 + hh * 32);
    uint* Xu = (uint*)(Xw + ln15 * 64 + hh * 32);
#pragma unroll
    for (int q4 = 0; q4 < 8; q4++) {
      float4 vv = s4[q4];
      union { uint u; h16 h[2]; } p0, p1;
      p0.h[0] = (h16)vv.x; p0.h[1] = (h16)vv.y;
      p1.h[0] = (h16)vv.z; p1.h[1] = (h16)vv.w;
      Xu[q4 * 2] = p0.u; Xu[q4 * 2 + 1] = p1.u;
    }
    lds_wait();
    v16h a0 = load_afrag(Xw, lane, 0), a1 = load_afrag(Xw, lane, 1);
#pragma unroll
    for (int nc = 0; nc < 4; nc++) {
      v8f acc = v8f{};
      acc = wmma16(a0, kfr[nc], acc);
      acc = wmma16(a1, kfr[4 + nc], acc);
#pragma unroll
      for (int i = 0; i < 8; i++)
        encK[(row0 + i + 8 * hh) * 64 + nc * 16 + ln15] = acc[i] + kbv[nc];
    }
#pragma unroll
    for (int nc = 0; nc < 4; nc++) {
      v8f acc = v8f{};
      acc = wmma16(a0, vfr[nc], acc);
      acc = wmma16(a1, vfr[4 + nc], acc);
#pragma unroll
      for (int i = 0; i < 8; i++)
        encV[(row0 + i + 8 * hh) * 64 + nc * 16 + ln15] = acc[i] + vbv[nc];
    }
  }
}

// ---------------- fused slot pass (K-logits / V-accumulate) -----------------
// MODE 0: logits[d] = (relu(mlp1(LN(encK+eg)))·w2q + b2q) * TSOFT
// MODE 1: partial[j] += sum_d wts[d]*relu(mlp1(LN(encV+eg)))[d,j]
// Grid: (b*S+s)*8 + nb ; block 256 = 8 waves x 4 tiles of 16 d-rows.
// mlp1 B-fragments live in registers (loaded once per wave from L2).

template <int MODE>
__global__ __launch_bounds__(256) void slot_pass(
    const float* __restrict__ enc, const float* __restrict__ posc,
    const h16* __restrict__ mfrag, const float* __restrict__ mlp1_b,
    const float* __restrict__ dense_w, const float* __restrict__ dense_b,
    const float* __restrict__ qN_g, const float* __restrict__ qN_b,
    const float* __restrict__ w2q, const float* __restrict__ b2q,
    const float* __restrict__ att, const float* __restrict__ invs,
    float* __restrict__ out) {
  __shared__ alignas(16) h16 X[8][1024];
  __shared__ float prm[6 * 64];                 // dw0,dw1,db,qg,qb,b1
  __shared__ float vred[8][64];
  int tid = threadIdx.x;
  if (tid < 64) {
    prm[tid]       = dense_w[tid];
    prm[64 + tid]  = dense_w[64 + tid];
    prm[128 + tid] = dense_b[tid];
    prm[192 + tid] = qN_g[tid];
    prm[256 + tid] = qN_b[tid];
    prm[320 + tid] = mlp1_b[tid];
  }
  int lane = tid & 31, w = tid >> 5, ln15 = lane & 15, hh = lane >> 4;
  v16h bf[8];
#pragma unroll
  for (int g = 0; g < 8; g++) bf[g] = load_bfrag(mfrag + g * 512, lane);
  __syncthreads();

  int nb = blockIdx.x & 7, bs = blockIdx.x >> 3;
  int b = bs / SS;
  int E0 = hh * 32;
  float px = posc[bs * 3], py = posc[bs * 3 + 1];
  float inv_ps = 1.0f / posc[bs * 3 + 2];

  float b1v[4], wqv[4] = {0, 0, 0, 0};
#pragma unroll
  for (int nc = 0; nc < 4; nc++) b1v[nc] = prm[320 + nc * 16 + ln15];
  float b2qv = 0.0f, invw = 0.0f;
  if (MODE == 0) {
#pragma unroll
    for (int nc = 0; nc < 4; nc++) wqv[nc] = w2q[bs * 64 + nc * 16 + ln15];
    b2qv = b2q[bs];
  } else {
    invw = invs[bs];
  }
  float vcol[4] = {0, 0, 0, 0};
  h16* Xw = X[w];

  for (int t = 0; t < 4; t++) {
    int d0 = nb * 512 + (w * 4 + t) * 16;
    int dr = d0 + ln15;
    float gx = -0.5f + (float)(dr & 63) * LINC;
    float gy = -0.5f + (float)(dr >> 6) * LINC;
    float rx = (gx - px) * inv_ps, ry = (gy - py) * inv_ps;
    const float4* s4 = (const float4*)(enc + ((size_t)b * DD + dr) * 64 + E0);
    float xv[32], sm = 0.0f, sq = 0.0f;
#pragma unroll
    for (int q4 = 0; q4 < 8; q4++) {
      float4 vv = s4[q4];
      int e = q4 * 4;
      xv[e + 0] = vv.x + rx * prm[E0 + e + 0] + ry * prm[64 + E0 + e + 0] + prm[128 + E0 + e + 0];
      xv[e + 1] = vv.y + rx * prm[E0 + e + 1] + ry * prm[64 + E0 + e + 1] + prm[128 + E0 + e + 1];
      xv[e + 2] = vv.z + rx * prm[E0 + e + 2] + ry * prm[64 + E0 + e + 2] + prm[128 + E0 + e + 2];
      xv[e + 3] = vv.w + rx * prm[E0 + e + 3] + ry * prm[64 + E0 + e + 3] + prm[128 + E0 + e + 3];
    }
#pragma unroll
    for (int e = 0; e < 32; e++) { sm += xv[e]; sq += xv[e] * xv[e]; }
    sm += __shfl_xor(sm, 16, 32);
    sq += __shfl_xor(sq, 16, 32);
    float mu = sm * (1.0f / 64.0f);
    float var = sq * (1.0f / 64.0f) - mu * mu;
    float rs = rsqrtf(var + 1e-5f);
    uint* Xu = (uint*)(Xw + ln15 * 64 + E0);
#pragma unroll
    for (int e2 = 0; e2 < 16; e2++) {
      float a0 = (xv[2 * e2] - mu) * rs * prm[192 + E0 + 2 * e2] + prm[256 + E0 + 2 * e2];
      float a1 = (xv[2 * e2 + 1] - mu) * rs * prm[192 + E0 + 2 * e2 + 1] + prm[256 + E0 + 2 * e2 + 1];
      union { uint u; h16 h[2]; } pk;
      pk.h[0] = (h16)a0; pk.h[1] = (h16)a1;
      Xu[e2] = pk.u;
    }
    lds_wait();
    v16h a0 = load_afrag(Xw, lane, 0), a1 = load_afrag(Xw, lane, 1);
    v8f accv[4];
#pragma unroll
    for (int nc = 0; nc < 4; nc++) {
      v8f acc = v8f{};
      acc = wmma16(a0, bf[nc], acc);
      acc = wmma16(a1, bf[4 + nc], acc);
      accv[nc] = acc;
    }
    if (MODE == 0) {
      float pr[8];
#pragma unroll
      for (int i = 0; i < 8; i++) {
        float t0 = fmaxf(accv[0][i] + b1v[0], 0.0f) * wqv[0];
        t0 += fmaxf(accv[1][i] + b1v[1], 0.0f) * wqv[1];
        t0 += fmaxf(accv[2][i] + b1v[2], 0.0f) * wqv[2];
        t0 += fmaxf(accv[3][i] + b1v[3], 0.0f) * wqv[3];
        pr[i] = t0;
      }
#pragma unroll
      for (int m = 1; m < 16; m <<= 1)
#pragma unroll
        for (int i = 0; i < 8; i++) pr[i] += __shfl_xor(pr[i], m, 32);
      if (ln15 == 0) {
        float* dst = out + (size_t)bs * DD + d0 + hh * 8;
#pragma unroll
        for (int i = 0; i < 8; i++) dst[i] = (pr[i] + b2qv) * 0.125f;
      }
    } else {
      const float* ap = att + (size_t)bs * DD + d0 + hh * 8;
      float wv[8];
#pragma unroll
      for (int i = 0; i < 8; i++) wv[i] = ap[i] * invw + 1e-8f;
#pragma unroll
      for (int nc = 0; nc < 4; nc++) {
        float s = 0.0f;
#pragma unroll
        for (int i = 0; i < 8; i++) s += wv[i] * fmaxf(accv[nc][i] + b1v[nc], 0.0f);
        vcol[nc] += s;
      }
    }
  }
  if (MODE == 1) {
#pragma unroll
    for (int nc = 0; nc < 4; nc++) vcol[nc] += __shfl_xor(vcol[nc], 16, 32);
    if (lane < 16)
#pragma unroll
      for (int nc = 0; nc < 4; nc++) vred[w][nc * 16 + lane] = vcol[nc];
    __syncthreads();
    if (tid < 64) {
      float s = 0.0f;
#pragma unroll
      for (int ww = 0; ww < 8; ww++) s += vred[ww][tid];
      out[((size_t)bs * 8 + nb) * 64 + tid] = s;   // deterministic partials
    }
  }
}

// ---------------- softmax over slots (axis=1) -------------------------------

__global__ __launch_bounds__(256) void softmax_slots(const float* __restrict__ logits,
                                                     float* __restrict__ att) {
  int idx = blockIdx.x * 256 + threadIdx.x;     // B*D
  int b = idx >> 12, d = idx & 4095;
  const float* lp = logits + (size_t)b * SS * DD + d;
  float l[SS], mx = -1e30f;
#pragma unroll
  for (int s = 0; s < SS; s++) { l[s] = lp[(size_t)s * DD]; mx = fmaxf(mx, l[s]); }
  float sum = 0.0f;
#pragma unroll
  for (int s = 0; s < SS; s++) { l[s] = expf(l[s] - mx); sum += l[s]; }
  float inv = 1.0f / sum;
  float* ap = att + (size_t)b * SS * DD + d;
#pragma unroll
  for (int s = 0; s < SS; s++) ap[(size_t)s * DD] = l[s] * inv;
}

// ---------------- per-(b,s) reductions + update_frames ----------------------

__device__ __forceinline__ float block_reduce256(float v, float* red, int tid) {
  __syncthreads();
  red[tid] = v; __syncthreads();
  for (int o = 128; o; o >>= 1) { if (tid < o) red[tid] += red[tid + o]; __syncthreads(); }
  return red[0];
}

__global__ __launch_bounds__(256) void reduce_bs(const float* __restrict__ att,
                                                 float* __restrict__ pos_next,
                                                 float* __restrict__ invs,
                                                 float* __restrict__ Sws) {
  __shared__ float red[256];
  int bs = blockIdx.x, tid = threadIdx.x;
  const float* ap = att + (size_t)bs * DD;
  float s0 = 0.0f;
  for (int d = tid; d < DD; d += 256) s0 += ap[d];
  float attsum = block_reduce256(s0, red, tid);
  float inv = 1.0f / attsum;
  float sw = 0, sx = 0, sy = 0, s2 = 0;
  for (int d = tid; d < DD; d += 256) {
    float wv = ap[d] * inv + 1e-8f;
    float gx = -0.5f + (float)(d & 63) * LINC;
    float gy = -0.5f + (float)(d >> 6) * LINC;
    sw += wv; sx += wv * gx; sy += wv * gy; s2 += wv * (gx * gx + gy * gy);
  }
  sw = block_reduce256(sw, red, tid);
  sx = block_reduce256(sx, red, tid);
  sy = block_reduce256(sy, red, tid);
  s2 = block_reduce256(s2, red, tid);
  if (tid == 0) {
    float np2 = sx * sx + sy * sy;
    float sc2 = s2 - 2.0f * np2 + np2 * sw;
    pos_next[bs * 3 + 0] = sx;
    pos_next[bs * 3 + 1] = sy;
    pos_next[bs * 3 + 2] = sqrtf(fmaxf(sc2, 0.0f));
    invs[bs] = inv; Sws[bs] = sw;
  }
}

// ---------------- GRU update + fold mlp2 into query -------------------------

__device__ __forceinline__ float block_reduce64(float v, float* red, int tid) {
  __syncthreads();
  red[tid] = v; __syncthreads();
  for (int o = 32; o; o >>= 1) { if (tid < o) red[tid] += red[tid + o]; __syncthreads(); }
  return red[0];
}

__device__ void q_tail(float qv, int e, int bs, float* red, float* QN,
                       const float* mlp2_w, const float* mlp2_b,
                       const float* qg, const float* qb,
                       float* w2q, float* b2q) {
  float sm = block_reduce64(qv, red, e);
  float mu = sm * (1.0f / 64.0f);
  float dv = qv - mu;
  float vr = block_reduce64(dv * dv, red, e) * (1.0f / 64.0f);
  float qn = dv * rsqrtf(vr + 1e-5f) * qg[e] + qb[e];
  QN[e] = qn;
  __syncthreads();
  float wq = 0.0f;
  for (int j = 0; j < 64; j++) wq += mlp2_w[e * 64 + j] * QN[j];
  w2q[bs * 64 + e] = wq;
  float bq = block_reduce64(mlp2_b[e] * qn, red, e);
  if (e == 0) b2q[bs] = bq;
}

__global__ __launch_bounds__(64) void gru_update(
    const float* __restrict__ partial, const float* __restrict__ Sws,
    const float* __restrict__ mlp2_w, const float* __restrict__ mlp2_b,
    const float* __restrict__ wih, const float* __restrict__ whh,
    const float* __restrict__ bih, const float* __restrict__ bhh,
    const float* __restrict__ qg, const float* __restrict__ qb,
    float* __restrict__ queries, float* __restrict__ w2q, float* __restrict__ b2q) {
  __shared__ float A[64], Xv[64], H[64], QN[64], red[64];
  int bs = blockIdx.x, e = threadIdx.x;
  {
    float a = 0.0f;
    for (int nb = 0; nb < 8; nb++) a += partial[((size_t)bs * 8 + nb) * 64 + e];
    A[e] = a;
  }
  H[e] = queries[bs * 64 + e];
  __syncthreads();
  float sw = Sws[bs];
  float u = sw * mlp2_b[e];
  for (int j = 0; j < 64; j++) u += A[j] * mlp2_w[j * 64 + e];
  Xv[e] = u;
  __syncthreads();
  float gr = bih[e], gz = bih[64 + e], gn = bih[128 + e];
  float hr = bhh[e], hz = bhh[64 + e], hn = bhh[128 + e];
  for (int j = 0; j < 64; j++) {
    float xj = Xv[j], hj = H[j];
    gr += xj * wih[e * 64 + j];        gz += xj * wih[(64 + e) * 64 + j];
    gn += xj * wih[(128 + e) * 64 + j];
    hr += hj * whh[e * 64 + j];        hz += hj * whh[(64 + e) * 64 + j];
    hn += hj * whh[(128 + e) * 64 + j];
  }
  float r = 1.0f / (1.0f + expf(-(gr + hr)));
  float z = 1.0f / (1.0f + expf(-(gz + hz)));
  float n = tanhf(gn + r * hn);
  float hnew = (1.0f - z) * n + z * H[e];
  queries[bs * 64 + e] = hnew;
  q_tail(hnew, e, bs, red, QN, mlp2_w, mlp2_b, qg, qb, w2q, b2q);
}

// ---------------- init: queries, pos_scale (with reference's quirks) --------

__global__ __launch_bounds__(64) void init_slots(
    const float* __restrict__ slot_noise, const float* __restrict__ pos_rand,
    const float* __restrict__ mu_, const float* __restrict__ logsig,
    const float* __restrict__ mlp2_w, const float* __restrict__ mlp2_b,
    const float* __restrict__ qg, const float* __restrict__ qb,
    float* __restrict__ queries, float* __restrict__ pos_cur,
    float* __restrict__ w2q, float* __restrict__ b2q) {
  __shared__ float QN[64], red[64];
  int bs = blockIdx.x, e = threadIdx.x, s = bs % SS;
  float q = mu_[e] + expf(logsig[e]) * slot_noise[bs * 64 + e];
  queries[bs * 64 + e] = q;
  if (e < 3) {
    float pr = pos_rand[bs * 3 + e];
    float v = (s < 2) ? pr - 0.5f : pr;
    if (s == SS - 1) v = (0.1f - 0.22360680f) * pr + 0.22360680f;  // (rlow-rhigh)*u+rhigh
    pos_cur[bs * 3 + e] = v;
  }
  q_tail(q, e, bs, red, QN, mlp2_w, mlp2_b, qg, qb, w2q, b2q);
}

// ---------------- final heads ----------------------------------------------

__global__ __launch_bounds__(64) void finalize_k(
    const float* __restrict__ queries, const float* __restrict__ nps,
    const float* __restrict__ a1w, const float* __restrict__ a1b,
    const float* __restrict__ a2w, const float* __restrict__ a2b,
    const float* __restrict__ f1w, const float* __restrict__ f1b,
    const float* __restrict__ f2w, const float* __restrict__ f2b,
    float* __restrict__ out_q, float* __restrict__ out_sf, float* __restrict__ out_alpha) {
  __shared__ float Q[64], T1[32], F1[64];
  int bs = blockIdx.x, e = threadIdx.x;
  Q[e] = queries[bs * 64 + e];
  __syncthreads();
  if (e < 32) {
    float a = a1b[e];
    for (int j = 0; j < 64; j++) a += Q[j] * a1w[j * 32 + e];
    T1[e] = fmaxf(a, 0.0f);
  }
  float f = f1b[e];
  for (int j = 0; j < 64; j++) f += Q[j] * f1w[j * 64 + e];
  F1[e] = fmaxf(f, 0.0f);
  __syncthreads();
  if (e == 0) {
    float aa = a2b[0];
    for (int k = 0; k < 32; k++) aa += T1[k] * a2w[k];
    out_alpha[bs] = 1.0f / (1.0f + expf(-aa));
  }
  if (e < 3) {
    float sf = f2b[e];
    for (int j = 0; j < 64; j++) sf += F1[j] * f2w[j * 3 + e];
    out_sf[bs * 3 + e] = sf + nps[bs * 3 + e];
  }
  out_q[bs * 64 + e] = Q[e];
}

// ---------------- host launcher --------------------------------------------

extern "C" void kernel_launch(void* const* d_in, const int* in_sizes, int n_in,
                              void* d_out, int out_size, void* d_ws, size_t ws_size,
                              hipStream_t stream) {
  (void)in_sizes; (void)n_in; (void)out_size; (void)ws_size;
  const float* data      = (const float*)d_in[0];
  const float* slotnoise = (const float*)d_in[1];
  const float* pos_rand  = (const float*)d_in[2];
  const float* conv0_w   = (const float*)d_in[3];
  const float* conv0_b   = (const float*)d_in[4];
  const float* conv1_w   = (const float*)d_in[5];
  const float* conv1_b   = (const float*)d_in[6];
  const float* conv2_w   = (const float*)d_in[7];
  const float* conv2_b   = (const float*)d_in[8];
  const float* dataN_g   = (const float*)d_in[9];
  const float* dataN_b   = (const float*)d_in[10];
  const float* queryN_g  = (const float*)d_in[11];
  const float* queryN_b  = (const float*)d_in[12];
  const float* toK_w     = (const float*)d_in[13];
  const float* toK_b     = (const float*)d_in[14];
  const float* toV_w     = (const float*)d_in[15];
  const float* toV_b     = (const float*)d_in[16];
  const float* dense_w   = (const float*)d_in[17];
  const float* dense_b   = (const float*)d_in[18];
  const float* mlp1_w    = (const float*)d_in[19];
  const float* mlp1_b    = (const float*)d_in[20];
  const float* mlp2_w    = (const float*)d_in[21];
  const float* mlp2_b    = (const float*)d_in[22];
  const float* gru_wih   = (const float*)d_in[23];
  const float* gru_whh   = (const float*)d_in[24];
  const float* gru_bih   = (const float*)d_in[25];
  const float* gru_bhh   = (const float*)d_in[26];
  const float* alpha1_w  = (const float*)d_in[27];
  const float* alpha1_b  = (const float*)d_in[28];
  const float* alpha2_w  = (const float*)d_in[29];
  const float* alpha2_b  = (const float*)d_in[30];
  const float* final1_w  = (const float*)d_in[31];
  const float* final1_b  = (const float*)d_in[32];
  const float* final2_w  = (const float*)d_in[33];
  const float* final2_b  = (const float*)d_in[34];
  const float* slots_mu  = (const float*)d_in[35];
  const float* slots_ls  = (const float*)d_in[36];

  char* ws = (char*)d_ws;
  auto alloc = [&](size_t bytes) -> char* {
    char* p = ws; ws += (bytes + 255) & ~(size_t)255; return p;
  };
  h16*   act0   = (h16*)  alloc((size_t)BB * DD * 64 * 2);
  h16*   act1   = (h16*)  alloc((size_t)BB * DD * 64 * 2);
  float* encLN  = (float*)alloc((size_t)BB * DD * 64 * 4);
  float* encK   = (float*)alloc((size_t)BB * DD * 64 * 4);
  float* encV   = (float*)alloc((size_t)BB * DD * 64 * 4);
  h16*   wf1    = (h16*)  alloc(102400 * 2);
  h16*   wf2    = (h16*)  alloc(102400 * 2);
  h16*   mf     = (h16*)  alloc(4096 * 2);
  h16*   kf     = (h16*)  alloc(4096 * 2);
  h16*   vf     = (h16*)  alloc(4096 * 2);
  float* logits = (float*)alloc((size_t)BB * SS * DD * 4);
  float* att_ws = (float*)alloc((size_t)BB * SS * DD * 4);
  float* queries= (float*)alloc((size_t)BB * SS * 64 * 4);
  float* posA   = (float*)alloc((size_t)BB * SS * 3 * 4);
  float* posB   = (float*)alloc((size_t)BB * SS * 3 * 4);
  float* w2q    = (float*)alloc((size_t)BB * SS * 64 * 4);
  float* b2q    = (float*)alloc((size_t)BB * SS * 4);
  float* partial= (float*)alloc((size_t)BB * SS * 8 * 64 * 4);
  float* invs   = (float*)alloc((size_t)BB * SS * 4);
  float* Sws    = (float*)alloc((size_t)BB * SS * 4);

  // weight pre-shuffles
  prep_frag64<<<16, 256, 0, stream>>>(mlp1_w, mf);
  prep_frag64<<<16, 256, 0, stream>>>(toK_w, kf);
  prep_frag64<<<16, 256, 0, stream>>>(toV_w, vf);
  prep_conv_frag<<<400, 256, 0, stream>>>(conv1_w, wf1);
  prep_conv_frag<<<400, 256, 0, stream>>>(conv2_w, wf2);

  // encoder
  conv0_k<<<256, 256, 0, stream>>>(data, conv0_w, conv0_b, act0);
  conv5x5<0><<<512, 256, 0, stream>>>(act0, wf1, conv1_b, nullptr, nullptr, act1, nullptr);
  conv5x5<1><<<512, 256, 0, stream>>>(act1, wf2, conv2_b, dataN_g, dataN_b, nullptr, encLN);
  enc_kv<<<128, 256, 0, stream>>>(encLN, kf, vf, toK_b, toV_b, encK, encV);

  init_slots<<<BB * SS, 64, 0, stream>>>(slotnoise, pos_rand, slots_mu, slots_ls,
                                         mlp2_w, mlp2_b, queryN_g, queryN_b,
                                         queries, posA, w2q, b2q);
  float* pcur = posA;
  float* pnext = posB;
  for (int it = 0; it < 3; ++it) {
    slot_pass<0><<<BB * SS * 8, 256, 0, stream>>>(encK, pcur, mf, mlp1_b, dense_w, dense_b,
                                                  queryN_g, queryN_b, w2q, b2q,
                                                  nullptr, nullptr, logits);
    softmax_slots<<<256, 256, 0, stream>>>(logits, att_ws);
    reduce_bs<<<BB * SS, 256, 0, stream>>>(att_ws, pnext, invs, Sws);
    slot_pass<1><<<BB * SS * 8, 256, 0, stream>>>(encV, pcur, mf, mlp1_b, dense_w, dense_b,
                                                  queryN_g, queryN_b, nullptr, nullptr,
                                                  att_ws, invs, partial);
    gru_update<<<BB * SS, 64, 0, stream>>>(partial, Sws, mlp2_w, mlp2_b,
                                           gru_wih, gru_whh, gru_bih, gru_bhh,
                                           queryN_g, queryN_b, queries, w2q, b2q);
    float* tmp = pcur; pcur = pnext; pnext = tmp;
  }

  float* out_q     = (float*)d_out;
  float* out_att   = out_q + BB * SS * 64;
  float* out_sf    = out_att + (size_t)BB * SS * DD;
  float* out_alpha = out_sf + BB * SS * 3;

  slot_pass<0><<<BB * SS * 8, 256, 0, stream>>>(encK, pcur, mf, mlp1_b, dense_w, dense_b,
                                                queryN_g, queryN_b, w2q, b2q,
                                                nullptr, nullptr, logits);
  softmax_slots<<<256, 256, 0, stream>>>(logits, out_att);
  reduce_bs<<<BB * SS, 256, 0, stream>>>(out_att, pnext, invs, Sws);
  finalize_k<<<BB * SS, 64, 0, stream>>>(queries, pnext, alpha1_w, alpha1_b, alpha2_w, alpha2_b,
                                         final1_w, final1_b, final2_w, final2_b,
                                         out_q, out_sf, out_alpha);
}
